// Seq_ViProj_Big_Attention_89232240541891
// MI455X (gfx1250) — matrix operand order repaired and checked by reading, not compiled, not run on hardware
//
#include <hip/hip_runtime.h>

// ---------------------------------------------------------------------------
// Seq_ViProj_Big_Attention for MI455X (gfx1250), bf16 WMMA pipeline.
// b=4, n=1024, m=8, qd=128, INNER=512, heads=8, d_eff=512 (head == modality).
// ---------------------------------------------------------------------------

typedef __attribute__((ext_vector_type(8)))  __bf16 v8bf;
typedef __attribute__((ext_vector_type(16))) __bf16 v16bf;
typedef __attribute__((ext_vector_type(8)))  float  v8f;

#define B_    4
#define N_    1024
#define M_    8
#define QD_   128
#define INNER 512
#define DEFF  512
#define FEAT  4096   // M_ * INNER == heads * DEFF

static __device__ __forceinline__ unsigned short f2bf(float f) {
  unsigned u = __float_as_uint(f);
  u += 0x7FFFu + ((u >> 16) & 1u);          // round-to-nearest-even
  return (unsigned short)(u >> 16);
}

// Build a 16-element bf16 fragment from two contiguous 16-byte chunks.
static __device__ __forceinline__ v16bf ldfrag(const unsigned short* p0,
                                               const unsigned short* p1) {
  v8bf lo = *(const v8bf*)p0;
  v8bf hi = *(const v8bf*)p1;
  v16bf r;
#pragma unroll
  for (int i = 0; i < 8; ++i) { r[i] = lo[i]; r[i + 8] = hi[i]; }
  return r;
}

static __device__ __forceinline__ v8f wmma_bf16(v16bf a, v16bf b, v8f c) {
  // (neg_a, A, neg_b, B, c_mod, C, reuse_a, reuse_b)
  return __builtin_amdgcn_wmma_f32_16x16x32_bf16(false, a, false, b,
                                                 (short)0, c, false, false);
}

// ---------------------------------------------------------------------------
// Pack kernels: f32 -> bf16 with GEMM-friendly layouts.
// ---------------------------------------------------------------------------

// x[b][n][m][ci] (f32) -> xbf[b][m][n][ci] (bf16): contiguous K for A-frags.
__global__ void pack_x_kernel(const float* __restrict__ x,
                              unsigned short* __restrict__ xbf) {
  int o = blockIdx.x * blockDim.x + threadIdx.x;
  if (o >= B_ * M_ * N_ * QD_) return;
  int ci = o & 127;
  int n  = (o >> 7) & 1023;
  int m  = (o >> 17) & 7;
  int b  = o >> 20;
  xbf[o] = f2bf(x[(((size_t)(b * N_ + n)) * M_ + m) * QD_ + ci]);
}

// W[co][ci][0][t] (f32, 512x128x1x5) -> wp[which][t][co][ci] (bf16).
__global__ void pack_w_kernel(const float* __restrict__ Wq,
                              const float* __restrict__ Wk,
                              const float* __restrict__ Wv,
                              unsigned short* __restrict__ wp) {
  int o = blockIdx.x * blockDim.x + threadIdx.x;
  if (o >= 3 * 5 * INNER * QD_) return;
  int ci = o & 127;
  int co = (o >> 7) & 511;
  int wt = o >> 16;                 // which*5 + t
  int which = wt / 5, t = wt - which * 5;
  const float* W = (which == 0) ? Wq : ((which == 1) ? Wk : Wv);
  wp[o] = f2bf(W[((size_t)co * QD_ + ci) * 5 + t]);
}

// Wo[out][in] (f32) -> bf16 (same layout: already B-fragment friendly).
__global__ void pack_wo_kernel(const float* __restrict__ Wo,
                               unsigned short* __restrict__ wob) {
  int o = blockIdx.x * blockDim.x + threadIdx.x;
  if (o >= 1024 * FEAT) return;
  wob[o] = f2bf(Wo[o]);
}

// ---------------------------------------------------------------------------
// Projection: conv1x5(reflect) as 5 shifted GEMMs, K=128 each, accumulated.
// Block: 128 threads (4 waves), tile 64(M=n) x 64(N=c_out).
// q,k stored [b][h][n][d]; v stored transposed [b][h][d][n].
// ---------------------------------------------------------------------------
__global__ __launch_bounds__(128)
void proj_kernel(const unsigned short* __restrict__ xbf,
                 const unsigned short* __restrict__ wp,
                 unsigned short* __restrict__ qb,
                 unsigned short* __restrict__ kb,
                 unsigned short* __restrict__ vt) {
  const int which = blockIdx.z;
  const int b = blockIdx.y >> 3, m = blockIdx.y & 7;
  const int mt = blockIdx.x & 15;   // 16 M-tiles of 64 rows (n)
  const int nt = blockIdx.x >> 4;   // 8  N-tiles of 64 cols (c_out)
  const int lane = threadIdx.x & 31;
  const int w    = threadIdx.x >> 5;
  const int ln = lane & 15, kg = lane >> 4;
  const int row0 = mt * 64 + w * 16;

  const unsigned short* xb = xbf + (size_t)((b * M_ + m) * N_) * QD_;
  const unsigned short* wb = wp + (size_t)which * 5 * INNER * QD_;

  v8f acc[4] = {};
#pragma unroll
  for (int t = 0; t < 5; ++t) {
    const unsigned short* wt = wb + (size_t)t * INNER * QD_;
    // reflect-padded source row for this lane's A-matrix row
    int idx = row0 + ln + t - 2;
    if (idx < 0) idx = -idx;
    if (idx > N_ - 1) idx = 2 * (N_ - 1) - idx;
    const unsigned short* pa_row = xb + (size_t)idx * QD_;
#pragma unroll
    for (int kc = 0; kc < 4; ++kc) {
      const int cib = kc * 32;
      v16bf af = ldfrag(pa_row + cib + kg * 8, pa_row + cib + 16 + kg * 8);
#pragma unroll
      for (int sub = 0; sub < 4; ++sub) {
        const int co = nt * 64 + sub * 16 + ln;
        const unsigned short* pb = wt + (size_t)co * QD_ + cib;
        v16bf bfr = ldfrag(pb + kg * 8, pb + 16 + kg * 8);
        acc[sub] = wmma_bf16(af, bfr, acc[sub]);
      }
    }
  }

  if (which < 2) {
    unsigned short* dst =
        (which == 0 ? qb : kb) + (size_t)((b * M_ + m) * N_) * INNER;
#pragma unroll
    for (int sub = 0; sub < 4; ++sub) {
      const int col = nt * 64 + sub * 16 + ln;
#pragma unroll
      for (int r = 0; r < 8; ++r) {
        const int row = row0 + r + (kg << 3);
        dst[(size_t)row * INNER + col] = f2bf(acc[sub][r]);
      }
    }
  } else {
    unsigned short* dst = vt + (size_t)(b * M_ + m) * INNER * N_;
#pragma unroll
    for (int sub = 0; sub < 4; ++sub) {
      const int col = nt * 64 + sub * 16 + ln;
#pragma unroll
      for (int r = 0; r < 8; ++r) {
        const int row = row0 + r + (kg << 3);
        dst[(size_t)col * N_ + row] = f2bf(acc[sub][r]);   // transposed
      }
    }
  }
}

// ---------------------------------------------------------------------------
// Flash attention. Block: 256 threads (8 waves), 16 query rows per block.
// Wave w owns d-slice [w*64, w*64+64). j-tiles of 64, 16 iterations.
// Cross-wave S reduction + P layout conversion via LDS.
// ---------------------------------------------------------------------------
__global__ __launch_bounds__(256)
void attn_kernel(const unsigned short* __restrict__ qb,
                 const unsigned short* __restrict__ kb,
                 const unsigned short* __restrict__ vt,
                 unsigned short* __restrict__ oa) {
  __shared__ float S[16][64];
  __shared__ __align__(16) unsigned short P[16][64];
  __shared__ float mrow[16], lrow[16], arow[16];

  const int b = blockIdx.y >> 3, h = blockIdx.y & 7;
  const int n0 = blockIdx.x * 16;
  const int tid = threadIdx.x;
  const int lane = tid & 31, w = tid >> 5;
  const int ln = lane & 15, kg = lane >> 4;

  const unsigned short* qh = qb + (size_t)((b * M_ + h) * N_) * DEFF;
  const unsigned short* kh = kb + (size_t)((b * M_ + h) * N_) * DEFF;
  const unsigned short* vh = vt + (size_t)(b * M_ + h) * DEFF * N_;

  if (tid < 16) { mrow[tid] = -1e30f; lrow[tid] = 0.f; }

  // Q fragments for this wave's d-slice, kept live for the whole j loop.
  v16bf qa[2];
  {
    const unsigned short* pq = qh + (size_t)(n0 + ln) * DEFF + w * 64;
#pragma unroll
    for (int kc = 0; kc < 2; ++kc)
      qa[kc] = ldfrag(pq + kc * 32 + kg * 8, pq + kc * 32 + 16 + kg * 8);
  }

  v8f o[4] = {};
  __syncthreads();

  for (int jt = 0; jt < 16; ++jt) {
    const int j0 = jt * 64;

    // zero S (1024 floats / 256 threads)
    float* Sf = &S[0][0];
#pragma unroll
    for (int i = 0; i < 4; ++i) Sf[tid * 4 + i] = 0.f;
    __syncthreads();

    // partial S over this wave's d-slice
    v8f s[4] = {};
#pragma unroll
    for (int sub = 0; sub < 4; ++sub) {
      const int j = j0 + sub * 16 + ln;
      const unsigned short* pk = kh + (size_t)j * DEFF + w * 64;
#pragma unroll
      for (int kc = 0; kc < 2; ++kc) {
        v16bf bk = ldfrag(pk + kc * 32 + kg * 8, pk + kc * 32 + 16 + kg * 8);
        s[sub] = wmma_bf16(qa[kc], bk, s[sub]);
      }
    }
    // reduce partial S across waves (ds_add_f32 atomics)
#pragma unroll
    for (int sub = 0; sub < 4; ++sub) {
      const int col = sub * 16 + ln;
#pragma unroll
      for (int r = 0; r < 8; ++r)
        atomicAdd(&S[r + (kg << 3)][col], s[sub][r]);
    }
    __syncthreads();

    // online softmax: one thread per query row
    if (tid < 16) {
      const int row = tid;
      float mold = mrow[row], mn = mold;
      for (int j = 0; j < 64; ++j) mn = fmaxf(mn, S[row][j] * 0.125f);
      const float alpha = __expf(mold - mn);
      float ls = 0.f;
      for (int j = 0; j < 64; ++j) {
        const float p = __expf(S[row][j] * 0.125f - mn);
        P[row][j] = f2bf(p);
        ls += p;
      }
      lrow[row] = lrow[row] * alpha + ls;
      mrow[row] = mn;
      arow[row] = alpha;
    }
    __syncthreads();

    // rescale accumulators by alpha of each C-layout row
#pragma unroll
    for (int r = 0; r < 8; ++r) {
      const float al = arow[r + (kg << 3)];
#pragma unroll
      for (int sub = 0; sub < 4; ++sub) o[sub][r] = o[sub][r] * al;
    }

    // out += P @ V^T  (P as A-frags from LDS, V pre-transposed [d][n])
#pragma unroll
    for (int kc = 0; kc < 2; ++kc) {
      v16bf pa = ldfrag(&P[ln][kc * 32 + kg * 8],
                        &P[ln][kc * 32 + 16 + kg * 8]);
#pragma unroll
      for (int sub = 0; sub < 4; ++sub) {
        const int d = w * 64 + sub * 16 + ln;
        const unsigned short* pv = vh + (size_t)d * N_ + j0 + kc * 32;
        v16bf bv = ldfrag(pv + kg * 8, pv + 16 + kg * 8);
        o[sub] = wmma_bf16(pa, bv, o[sub]);
      }
    }
    __syncthreads();   // protect P/S before next iteration
  }

  // epilogue: divide by l, store bf16 into [b*n][h*512 + d] (A-layout for GEMM3)
#pragma unroll
  for (int r = 0; r < 8; ++r) {
    const float inv = 1.f / lrow[r + (kg << 3)];
    const int row = n0 + r + (kg << 3);
    unsigned short* dst = oa + ((size_t)(b * N_) + row) * FEAT;
#pragma unroll
    for (int sub = 0; sub < 4; ++sub) {
      const int col = h * DEFF + w * 64 + sub * 16 + ln;
      dst[col] = f2bf(o[sub][r] * inv);
    }
  }
}

// ---------------------------------------------------------------------------
// Output linear: (4096 x 4096) @ Wo^T (4096 x 1024) + bo, f32 out.
// Block: 128 threads (4 waves), 64x64 tile.
// ---------------------------------------------------------------------------
__global__ __launch_bounds__(128)
void out_kernel(const unsigned short* __restrict__ oa,
                const unsigned short* __restrict__ wob,
                const float* __restrict__ bo,
                float* __restrict__ y) {
  const int mt = blockIdx.x & 63;   // 64 M-tiles
  const int nt = blockIdx.x >> 6;   // 16 N-tiles
  const int lane = threadIdx.x & 31, w = threadIdx.x >> 5;
  const int ln = lane & 15, kg = lane >> 4;
  const int row0 = mt * 64 + w * 16;

  v8f acc[4] = {};
  const unsigned short* pa = oa + (size_t)(row0 + ln) * FEAT;
  for (int kbase = 0; kbase < FEAT; kbase += 32) {
    if (kbase + 512 < FEAT)
      __builtin_prefetch(pa + kbase + 512, 0, 1);   // global_prefetch_b8
    v16bf af = ldfrag(pa + kbase + kg * 8, pa + kbase + 16 + kg * 8);
#pragma unroll
    for (int sub = 0; sub < 4; ++sub) {
      const int col = nt * 64 + sub * 16 + ln;
      const unsigned short* pb = wob + (size_t)col * FEAT + kbase;
      v16bf bfr = ldfrag(pb + kg * 8, pb + 16 + kg * 8);
      acc[sub] = wmma_bf16(af, bfr, acc[sub]);
    }
  }
#pragma unroll
  for (int sub = 0; sub < 4; ++sub) {
    const int col = nt * 64 + sub * 16 + ln;
    const float bias = bo[col];
#pragma unroll
    for (int r = 0; r < 8; ++r) {
      const int row = row0 + r + (kg << 3);
      y[(size_t)row * 1024 + col] = acc[sub][r] + bias;
    }
  }
}

// ---------------------------------------------------------------------------
// Host launcher.
// ---------------------------------------------------------------------------
extern "C" void kernel_launch(void* const* d_in, const int* in_sizes, int n_in,
                              void* d_out, int out_size, void* d_ws, size_t ws_size,
                              hipStream_t stream) {
  const float* x  = (const float*)d_in[0];
  const float* Wq = (const float*)d_in[1];
  const float* Wk = (const float*)d_in[2];
  const float* Wv = (const float*)d_in[3];
  const float* Wo = (const float*)d_in[4];
  const float* bo = (const float*)d_in[5];
  float* y = (float*)d_out;

  char* ws = (char*)d_ws;
  size_t off = 0;
  unsigned short* xbf = (unsigned short*)(ws + off); off += (size_t)B_ * M_ * N_ * QD_ * 2;     //  8 MB
  unsigned short* wp  = (unsigned short*)(ws + off); off += (size_t)3 * 5 * INNER * QD_ * 2;    // ~2 MB
  unsigned short* wob = (unsigned short*)(ws + off); off += (size_t)1024 * FEAT * 2;            //  8 MB
  unsigned short* qb  = (unsigned short*)(ws + off); off += (size_t)B_ * M_ * N_ * DEFF * 2;    // 32 MB
  unsigned short* kb  = (unsigned short*)(ws + off); off += (size_t)B_ * M_ * N_ * DEFF * 2;    // 32 MB
  unsigned short* vt  = (unsigned short*)(ws + off); off += (size_t)B_ * M_ * DEFF * N_ * 2;    // 32 MB
  unsigned short* oa  = (unsigned short*)(ws + off); off += (size_t)B_ * N_ * FEAT * 2;         // 32 MB

  // 1) pack inputs to bf16
  pack_x_kernel<<<(B_ * M_ * N_ * QD_) / 256, 256, 0, stream>>>(x, xbf);
  pack_w_kernel<<<(3 * 5 * INNER * QD_ + 255) / 256, 256, 0, stream>>>(Wq, Wk, Wv, wp);
  pack_wo_kernel<<<(1024 * FEAT) / 256, 256, 0, stream>>>(Wo, wob);

  // 2) conv1x5 projections as WMMA GEMMs (q, k row-major; v transposed)
  proj_kernel<<<dim3(128, B_ * M_, 3), 128, 0, stream>>>(xbf, wp, qb, kb, vt);

  // 3) flash attention (16 query rows / block, 32 (b,h) pairs)
  attn_kernel<<<dim3(N_ / 16, B_ * M_), 256, 0, stream>>>(qb, kb, vt, oa);

  // 4) output linear + bias
  out_kernel<<<dim3(64 * 16), 128, 0, stream>>>(oa, wob, bo, y);
}
